// TensorProduct_12713103196326
// MI455X (gfx1250) — matrix operand order repaired and checked by reading, not compiled, not run on hardware
//
#include <hip/hip_runtime.h>
#include <hip/hip_bf16.h>
#include <stdint.h>

typedef __attribute__((ext_vector_type(16))) _Float16 v16h;
typedef __attribute__((ext_vector_type(2)))  _Float16 h2;
typedef __attribute__((ext_vector_type(8)))  float    v8f;

#define DIM_IN  16
#define NCH     64
#define DIM_OUT 99
#define NPAIR   256   // 16*16 (mu1,mu2) pairs
#define NMT     7     // ceil(99/16) output M-tiles
#define NKC     8     // 256/32 K-chunks

// ---------------------------------------------------------------------------
// Init kernel: compute e3nn real-basis Wigner-3j CG coefficients (double,
// Racah formula + complex->real basis change, unit Frobenius norm, *sqrt(2l3+1))
// and pack them into the exact WMMA A-fragment layout (f16) in workspace.
// ---------------------------------------------------------------------------

__device__ __forceinline__ double facd(int n) {
    const double t[11] = {1., 1., 2., 6., 24., 120., 720., 5040., 40320., 362880., 3628800.};
    return t[n];
}

__device__ double su2_cg(int j1, int m1, int j2, int m2, int j3, int m3) {
    if (m3 != m1 + m2) return 0.0;
    int vmin = -j1 + j2 + m3;
    if (-j1 + m1 > vmin) vmin = -j1 + m1;
    if (0 > vmin) vmin = 0;
    int vmax = j2 + j3 + m1;
    if (j3 - j1 + j2 < vmax) vmax = j3 - j1 + j2;
    if (j3 + m3 < vmax) vmax = j3 + m3;
    double C = (2.0 * j3 + 1.0) *
               (facd(j3 + j1 - j2) * facd(j3 - j1 + j2) * facd(j1 + j2 - j3) *
                facd(j3 + m3) * facd(j3 - m3)) /
               (facd(j1 + j2 + j3 + 1) * facd(j1 - m1) * facd(j1 + m1) *
                facd(j2 - m2) * facd(j2 + m2));
    double S = 0.0;
    for (int v = vmin; v <= vmax; v++) {
        double sgn = ((v + j2 + m2) & 1) ? -1.0 : 1.0;
        S += sgn * (facd(j2 + j3 + m1 - v) * facd(j1 - m1 + v)) /
             (facd(v) * facd(j3 - j1 + j2 - v) * facd(j3 + m3 - v) * facd(v + j1 - j2 - m3));
    }
    return sqrt(C) * S;
}

// q_real_to_complex (matches e3nn, incl. (-i)^l phase)
__device__ void build_q(int l, double qr[7][7], double qi[7][7]) {
    for (int a = 0; a < 7; a++)
        for (int b = 0; b < 7; b++) { qr[a][b] = 0.0; qi[a][b] = 0.0; }
    const double s = 0.70710678118654752440;
    for (int m = -l; m < 0; m++) {
        qr[l + m][l - m] = s;     // 1/sqrt2 at col l+|m|
        qi[l + m][l + m] = -s;    // -i/sqrt2 at col l-|m|
    }
    qr[l][l] = 1.0;
    for (int m = 1; m <= l; m++) {
        double sg = (m & 1) ? -1.0 : 1.0;
        qr[l + m][l + m] = sg * s;
        qi[l + m][l - m] = sg * s;
    }
    int ph = l & 3; // multiply by (-i)^l : 0->1, 1->-i, 2->-1, 3->i
    for (int a = 0; a < 2 * l + 1; a++)
        for (int b = 0; b < 2 * l + 1; b++) {
            double re = qr[a][b], im = qi[a][b];
            if (ph == 1)      { qr[a][b] =  im; qi[a][b] = -re; }
            else if (ph == 2) { qr[a][b] = -re; qi[a][b] = -im; }
            else if (ph == 3) { qr[a][b] = -im; qi[a][b] =  re; }
        }
}

__global__ void cg_init_kernel(float* __restrict__ dense, _Float16* __restrict__ atab) {
    // phase 1: zero dense CG matrix [99][256]
    for (int i = threadIdx.x; i < DIM_OUT * NPAIR; i += blockDim.x) dense[i] = 0.f;
    __syncthreads();

    // phase 2: 23 paths, one thread each; output row ranges are disjoint.
    if (threadIdx.x < 23) {
        const int P[23][6] = {
            {0,0,0, 0,0, 0}, {0,1,1, 0,1, 1}, {0,2,2, 0,4, 4}, {0,3,3, 0,9, 9},
            {1,0,1, 1,0,16}, {1,1,0, 1,1,19}, {1,1,2, 1,1,20}, {1,2,1, 1,4,25},
            {1,2,3, 1,4,28}, {1,3,2, 1,9,35},
            {2,0,2, 4,0,40}, {2,1,1, 4,1,45}, {2,1,3, 4,1,48}, {2,2,0, 4,4,55},
            {2,2,2, 4,4,56}, {2,3,1, 4,9,61}, {2,3,3, 4,9,64},
            {3,0,3, 9,0,71}, {3,1,2, 9,1,78}, {3,2,1, 9,4,83}, {3,2,3, 9,4,86},
            {3,3,0, 9,9,93}, {3,3,2, 9,9,94}
        };
        int t  = threadIdx.x;
        int l1 = P[t][0], l2 = P[t][1], l3 = P[t][2];
        int o1 = P[t][3], o2 = P[t][4], o3 = P[t][5];
        int n1 = 2 * l1 + 1, n2 = 2 * l2 + 1, n3 = 2 * l3 + 1;

        double q1r[7][7], q1i[7][7], q2r[7][7], q2i[7][7], q3r[7][7], q3i[7][7];
        build_q(l1, q1r, q1i);
        build_q(l2, q2r, q2i);
        build_q(l3, q3r, q3i);

        double blk[7][7][7];
        double ss = 0.0;
        // einsum('ij,kl,mn,ikn->jlm', Q1, Q2, conj(Q3.T), C) ; keep real part
        for (int a = 0; a < n1; a++)
            for (int b = 0; b < n2; b++)
                for (int c = 0; c < n3; c++) {
                    double accr = 0.0;
                    for (int i = 0; i < n1; i++) {
                        double c1r = q1r[i][a], c1i = q1i[i][a];
                        if (c1r == 0.0 && c1i == 0.0) continue;
                        for (int k = 0; k < n2; k++) {
                            double t2r = c1r * q2r[k][b] - c1i * q2i[k][b];
                            double t2i = c1r * q2i[k][b] + c1i * q2r[k][b];
                            if (t2r == 0.0 && t2i == 0.0) continue;
                            for (int n = 0; n < n3; n++) {
                                double cgv = su2_cg(l1, i - l1, l2, k - l2, l3, n - l3);
                                if (cgv == 0.0) continue;
                                double c3r = q3r[n][c], c3i = -q3i[n][c]; // conj(Q3[n][c])
                                accr += (t2r * c3r - t2i * c3i) * cgv;
                            }
                        }
                    }
                    blk[a][b][c] = accr;
                    ss += accr * accr;
                }
        double scale = sqrt((double)n3) / sqrt(ss); // unit Frobenius * sqrt(2l3+1)
        for (int a = 0; a < n1; a++)
            for (int b = 0; b < n2; b++)
                for (int c = 0; c < n3; c++)
                    dense[(o3 + c) * NPAIR + (o1 + a) * 16 + (o2 + b)] =
                        (float)(blk[a][b][c] * scale);
    }
    __syncthreads();

    // phase 3: pack into A-fragment layout (16-bit A, 16x32 per fragment):
    // lanes 0-15: M=lane,   K = e
    // lanes 16-31:M=lane-16,K = 16+e
    for (int idx = threadIdx.x; idx < NMT * NKC * 32 * 16; idx += blockDim.x) {
        int e    = idx & 15;
        int lane = (idx >> 4) & 31;
        int kc   = (idx >> 9) & 7;
        int mt   = idx >> 12;
        int hi   = lane >> 4;
        int row  = mt * 16 + (lane & 15);
        int kl   = e + hi * 16;
        int p    = kc * 32 + kl;
        float v  = (row < DIM_OUT) ? dense[row * NPAIR + p] : 0.f;
        atab[idx] = (_Float16)v;
    }
}

// ---------------------------------------------------------------------------
// Main kernel: per batch element, out[99,64] = CG[99,256] x Z[256,64],
// Z[p,c] = x[p>>4,c]*y[p&15,c], via v_wmma_f32_16x16x32_f16.
// Block = 128 threads = 4 waves; wave w owns channels [16w,16w+16).
// B-fragment for K-chunk kc is lane-local: yh_vec * splat(xh[2kc+laneHi]).
// ---------------------------------------------------------------------------

__global__ __launch_bounds__(128)
void tp_wmma_kernel(const float* __restrict__ x, const float* __restrict__ y,
                    const _Float16* __restrict__ atab, float* __restrict__ out) {
    int b    = blockIdx.x;
    int lane = threadIdx.x & 31;
    int wv   = threadIdx.x >> 5;      // N-tile (channel group of 16)
    int col  = lane & 15;
    int hi   = lane >> 4;
    int c    = wv * 16 + col;

    const float* xb = x + (size_t)b * DIM_IN * NCH + c;
    const float* yb = y + (size_t)b * DIM_IN * NCH + c;

    // Load x[0..15,c], y[0..15,c]; convert to f16 pairs.
    uint32_t xu[8];
    h2       yv[8];
#pragma unroll
    for (int t = 0; t < 8; t++) {
        float x0 = xb[(2 * t) * NCH], x1 = xb[(2 * t + 1) * NCH];
        float y0 = yb[(2 * t) * NCH], y1 = yb[(2 * t + 1) * NCH];
        h2 xh; xh.x = (_Float16)x0; xh.y = (_Float16)x1;
        h2 yh; yh.x = (_Float16)y0; yh.y = (_Float16)y1;
        xu[t] = __builtin_bit_cast(uint32_t, xh);
        yv[t] = yh;
    }

    union V16 { h2 p[8]; v16h v; };

    // Precompute all 8 B fragments (K-chunks), lane-local, no LDS / shuffles.
    v16h bfrag[8];
#pragma unroll
    for (int kc = 0; kc < 8; kc++) {
        uint32_t u    = xu[kc];
        uint32_t hsel = hi ? (u >> 16) : (u & 0xffffu);
        uint32_t sp   = hsel | (hsel << 16);
        h2 xs = __builtin_bit_cast(h2, sp);
        V16 t;
#pragma unroll
        for (int j = 0; j < 8; j++) t.p[j] = yv[j] * xs;   // v_pk_mul_f16
        bfrag[kc] = t.v;
    }

    float* ob = out + (size_t)b * DIM_OUT * NCH + c;

#pragma unroll
    for (int mt = 0; mt < NMT; mt++) {
        v8f acc = {};
#pragma unroll
        for (int kc = 0; kc < NKC; kc++) {
            const v16h* ap =
                (const v16h*)(atab + ((((size_t)mt * NKC + kc) * 32 + lane) << 4));
            v16h a = *ap;  // 32B A-fragment load (WGP$-resident table)
            acc = __builtin_amdgcn_wmma_f32_16x16x32_f16(
                false, a, false, bfrag[kc], (short)0, acc, false, false);
        }
        // D layout: VGPR r -> (M = mt*16 + r + 8*hi, N = col)
        int rbase = mt * 16 + hi * 8;
#pragma unroll
        for (int r = 0; r < 8; r++) {
            int row = rbase + r;
            if (row < DIM_OUT) ob[(size_t)row * NCH] = acc[r];
        }
    }
}

// ---------------------------------------------------------------------------

extern "C" void kernel_launch(void* const* d_in, const int* in_sizes, int n_in,
                              void* d_out, int out_size, void* d_ws, size_t ws_size,
                              hipStream_t stream) {
    (void)n_in; (void)out_size; (void)ws_size;
    const float* x = (const float*)d_in[0];
    const float* y = (const float*)d_in[1];
    float*       out = (float*)d_out;

    // ws layout: [0, 101376) dense CG f32 [99][256]; then A-frag f16 table (57344 B)
    float*    dense = (float*)d_ws;
    _Float16* atab  = (_Float16*)((char*)d_ws + (size_t)DIM_OUT * NPAIR * 4);

    int B = in_sizes[0] / (DIM_IN * NCH);

    cg_init_kernel<<<1, 256, 0, stream>>>(dense, atab);
    tp_wmma_kernel<<<B, 128, 0, stream>>>(x, y, atab, out);
}
